// TransAttentionLayer_65206193488148
// MI455X (gfx1250) — compile-verified, matrix operand order checked by reading, not run on hardware
//
#include <hip/hip_runtime.h>

// bf16 WMMA operand/accumulator vector types (probe-confirmed signatures)
typedef __attribute__((ext_vector_type(16))) __bf16 v16bf;
typedef __attribute__((ext_vector_type(8)))  __bf16 v8bf;
typedef __attribute__((ext_vector_type(8)))  float  v8f;

#define NN   4096   // n_nodes
#define DD   128    // in_features
#define AA   128    // attention dim
#define HH   4      // heads
// (1/sqrt(32)) * log2(e): fold softmax scale + base-2 conversion into one mul
#define SCL2 0.25503508f

// Load a v16bf WMMA operand for one lane: two contiguous 8x bf16 (16B) runs at
// base and base+16, matching the 16-bit A/B VGPR layout (K split 0..7/16..23
// for lanes 0-15, 8..15/24..31 for lanes 16-31 — caller passes half-adjusted base).
__device__ __forceinline__ v16bf load_ab(const __bf16* base) {
    v8bf lo = *(const v8bf*)(base);
    v8bf hi = *(const v8bf*)(base + 16);
    return __builtin_shufflevector(lo, hi, 0,1,2,3,4,5,6,7,8,9,10,11,12,13,14,15);
}

// ---------------------------------------------------------------------------
// Kernel 1: convert x -> bf16 (both [d][N] and transposed [N][d]) and
// transpose+convert Qw/Kw so WMMA B-operand K-runs are contiguous.
// ---------------------------------------------------------------------------
__global__ __launch_bounds__(256) void prep_kernel(
    const float* __restrict__ x, const float* __restrict__ Qw,
    const float* __restrict__ Kw,
    __bf16* __restrict__ xt, __bf16* __restrict__ xbf,
    __bf16* __restrict__ wtq, __bf16* __restrict__ wtk) {
    int tid = blockIdx.x * 256 + threadIdx.x;        // 0 .. 128*4096-1
    int k = tid >> 12;                               // d index
    int n = tid & 4095;                              // node index
    float v = x[tid];                                // x is [d][N] row-major
    xbf[tid]          = (__bf16)v;                   // [d][N] bf16
    xt[n * DD + k]    = (__bf16)v;                   // [N][d] bf16
    if (tid < DD * AA) {
        int kk = tid >> 7, a = tid & 127;            // Qw/Kw are [d][A]
        wtq[a * DD + kk] = (__bf16)Qw[tid];          // [A][d]
        wtk[a * DD + kk] = (__bf16)Kw[tid];
    }
}

// ---------------------------------------------------------------------------
// Kernel 2: Q = x^T @ Qw + Qb, K = x^T @ Kw + Kb  ->  bf16 [N][A]
// One wave per 16x16 output tile; K-dim 128 = 4 chained WMMA(16x16x32).
// ---------------------------------------------------------------------------
__global__ __launch_bounds__(256) void proj_kernel(
    const __bf16* __restrict__ xt,
    const __bf16* __restrict__ wtq, const __bf16* __restrict__ wtk,
    const float* __restrict__ Qb, const float* __restrict__ Kb,
    __bf16* __restrict__ qbf, __bf16* __restrict__ kbf) {
    int w    = blockIdx.x * 8 + (threadIdx.x >> 5);  // 0..4095
    int l    = threadIdx.x & 31;
    int lm   = l & 15, half = l >> 4;
    int sel  = w >> 11;                              // 0 = Q, 1 = K
    int rem  = w & 2047;
    int n0   = (rem >> 3) * 16;
    int a0   = (rem & 7) * 16;
    const __bf16* wt   = sel ? wtk : wtq;
    const float*  bias = sel ? Kb  : Qb;
    __bf16*       dst  = sel ? kbf : qbf;

    v8f acc = {};
    #pragma unroll
    for (int kk = 0; kk < 4; ++kk) {
        v16bf A = load_ab(xt + (n0 + lm) * DD + kk * 32 + half * 8);
        v16bf B = load_ab(wt + (a0 + lm) * DD + kk * 32 + half * 8);
        acc = __builtin_amdgcn_wmma_f32_16x16x32_bf16(false, A, false, B,
                                                      (short)0, acc, false, false);
    }
    float b = bias[a0 + lm];                         // C-layout: lane == column
    #pragma unroll
    for (int i = 0; i < 8; ++i) {
        int n = n0 + half * 8 + i;                   // C-layout: VGPR i == row
        dst[n * AA + a0 + lm] = (__bf16)(acc[i] + b);
    }
}

// ---------------------------------------------------------------------------
// Kernel 3: softmax row stats. One wave per (head, 16-row tile of n).
// Pass 1: rowmax over all m (WMMA + v_max, single cross-lane reduce at end).
// Pass 2: rowsum of exp2(SCL2*s - c). Stores the single combined constant
//   cc[h][n] = SCL2*rowmax + log2(H * rowsum)
// so the consumer computes P with ONE load and ONE exp2:
//   P = exp2(SCL2*s - cc)  ==  exp(scale*(s-max)) / (H*sum).
// ---------------------------------------------------------------------------
__global__ __launch_bounds__(256) void stats_kernel(
    const __bf16* __restrict__ qbf, const __bf16* __restrict__ kbf,
    float* __restrict__ cc) {
    int w  = blockIdx.x * 8 + (threadIdx.x >> 5);    // 0..1023
    int l  = threadIdx.x & 31;
    int lm = l & 15, half = l >> 4;
    int h  = w >> 8;                                 // head
    int n0 = (w & 255) * 16;

    v16bf A = load_ab(qbf + (n0 + lm) * AA + h * 32 + half * 8);
    v8f z = {};
    float mx[8];
    #pragma unroll
    for (int i = 0; i < 8; ++i) mx[i] = -3.0e38f;

    for (int mt = 0; mt < NN / 16; ++mt) {
        v16bf B = load_ab(kbf + (mt * 16 + lm) * AA + h * 32 + half * 8);
        v8f s = __builtin_amdgcn_wmma_f32_16x16x32_bf16(false, A, false, B,
                                                        (short)0, z, false, false);
        #pragma unroll
        for (int i = 0; i < 8; ++i) mx[i] = fmaxf(mx[i], s[i]);
    }
    // reduce across the 16 lanes holding one row (stay inside each half-wave)
    #pragma unroll
    for (int i = 0; i < 8; ++i) {
        #pragma unroll
        for (int m = 1; m < 16; m <<= 1)
            mx[i] = fmaxf(mx[i], __shfl_xor(mx[i], m, 16));
        mx[i] *= SCL2;                               // pre-scaled log2-domain max
    }

    float sm[8];
    #pragma unroll
    for (int i = 0; i < 8; ++i) sm[i] = 0.0f;
    for (int mt = 0; mt < NN / 16; ++mt) {
        v16bf B = load_ab(kbf + (mt * 16 + lm) * AA + h * 32 + half * 8);
        v8f s = __builtin_amdgcn_wmma_f32_16x16x32_bf16(false, A, false, B,
                                                        (short)0, z, false, false);
        #pragma unroll
        for (int i = 0; i < 8; ++i) sm[i] += exp2f(SCL2 * s[i] - mx[i]);
    }
    #pragma unroll
    for (int i = 0; i < 8; ++i) {
        #pragma unroll
        for (int m = 1; m < 16; m <<= 1)
            sm[i] += __shfl_xor(sm[i], m, 16);
    }
    if (lm == 0) {
        #pragma unroll
        for (int i = 0; i < 8; ++i) {
            int n = n0 + half * 8 + i;
            // combined exponent offset: SCL2*max + log2(H*sum)  (v_log_f32)
            cc[h * NN + n] = mx[i] + log2f((float)HH * sm[i]);
        }
    }
}

// ---------------------------------------------------------------------------
// Kernel 4: out[:, m-stripe] = sum_n x[:, n] * P[n, m].
// Block = 16 m-columns x all 128 d-rows; 8 waves each own one 16x16 out tile.
// Per 128-wide n chunk: each wave recomputes its 4 head score tiles (WMMA),
// forms P = sum_h exp2(SCL2*s - cc_h) in fp32 (one load + one exp2 per term),
// stages P^T as bf16 in LDS, then 4 accumulating WMMAs of x_bf16 (A) x P (B).
// ---------------------------------------------------------------------------
__global__ __launch_bounds__(256) void out_kernel(
    const __bf16* __restrict__ qbf, const __bf16* __restrict__ kbf,
    const __bf16* __restrict__ xbf,
    const float* __restrict__ cc,
    float* __restrict__ out) {
    __shared__ __align__(16) __bf16 Plds[16][128];   // [m][n_local], 4 KB

    int m0 = blockIdx.x * 16;
    int w  = threadIdx.x >> 5;                       // wave id 0..7
    int l  = threadIdx.x & 31;
    int lm = l & 15, half = l >> 4;

    v16bf Bk[HH];                                    // K tiles, fixed per block
    #pragma unroll
    for (int h = 0; h < HH; ++h)
        Bk[h] = load_ab(kbf + (m0 + lm) * AA + h * 32 + half * 8);

    v8f z   = {};
    v8f acc = {};

    for (int nc = 0; nc < NN; nc += 128) {
        float p[8];
        #pragma unroll
        for (int i = 0; i < 8; ++i) p[i] = 0.0f;
        #pragma unroll
        for (int h = 0; h < HH; ++h) {
            v16bf Aq = load_ab(qbf + (nc + w * 16 + lm) * AA + h * 32 + half * 8);
            v8f s = __builtin_amdgcn_wmma_f32_16x16x32_bf16(false, Aq, false, Bk[h],
                                                            (short)0, z, false, false);
            #pragma unroll
            for (int i = 0; i < 8; ++i) {
                int n = nc + w * 16 + half * 8 + i;  // score row (query index)
                p[i] += exp2f(SCL2 * s[i] - cc[h * NN + n]);
            }
        }
        #pragma unroll
        for (int i = 0; i < 8; ++i)                  // store P^T: [m][n_local]
            Plds[lm][w * 16 + half * 8 + i] = (__bf16)p[i];
        __syncthreads();

        #pragma unroll
        for (int kk = 0; kk < 4; ++kk) {
            v16bf Ax = load_ab(xbf + (w * 16 + lm) * NN + nc + kk * 32 + half * 8);
            v16bf Bp = load_ab(&Plds[lm][kk * 32 + half * 8]);
            acc = __builtin_amdgcn_wmma_f32_16x16x32_bf16(false, Ax, false, Bp,
                                                          (short)0, acc, false, false);
        }
        __syncthreads();                             // protect Plds before rewrite
    }

    #pragma unroll
    for (int i = 0; i < 8; ++i)
        out[(w * 16 + half * 8 + i) * NN + m0 + lm] = acc[i];
}

// ---------------------------------------------------------------------------
extern "C" void kernel_launch(void* const* d_in, const int* in_sizes, int n_in,
                              void* d_out, int out_size, void* d_ws, size_t ws_size,
                              hipStream_t stream) {
    (void)in_sizes; (void)n_in; (void)out_size; (void)ws_size;
    const float* x  = (const float*)d_in[0];
    const float* Qw = (const float*)d_in[1];
    const float* Qb = (const float*)d_in[2];
    const float* Kw = (const float*)d_in[3];
    const float* Kb = (const float*)d_in[4];
    // d_in[5] (Vw) and d_in[6] (Vb) are unused, as in the reference.
    float* out = (float*)d_out;

    char* ws = (char*)d_ws;                          // ~4.2 MB used
    __bf16* xt  = (__bf16*)(ws);                     // [N][d]   1 MB
    __bf16* xbf = (__bf16*)(ws + (1 << 20));         // [d][N]   1 MB
    __bf16* qbf = (__bf16*)(ws + (2 << 20));         // [N][A]   1 MB
    __bf16* kbf = (__bf16*)(ws + (3 << 20));         // [N][A]   1 MB
    __bf16* wtq = (__bf16*)(ws + (4 << 20));         // [A][d]   32 KB
    __bf16* wtk = (__bf16*)(ws + (4 << 20) + 32768); // [A][d]   32 KB
    float*  cc  = (float*)(ws + (4 << 20) + 65536);  // [H][N]   64 KB

    prep_kernel <<<(DD * NN) / 256, 256, 0, stream>>>(x, Qw, Kw, xt, xbf, wtq, wtk);
    proj_kernel <<<512, 256, 0, stream>>>(xt, wtq, wtk, Qb, Kb, qbf, kbf);
    stats_kernel<<<128, 256, 0, stream>>>(qbf, kbf, cc);
    out_kernel  <<<NN / 16, 256, 0, stream>>>(qbf, kbf, xbf, cc, out);
}
